// MRG_Graph_77927886618971
// MI455X (gfx1250) — matrix operand order repaired and checked by reading, not compiled, not run on hardware
//
#include <hip/hip_runtime.h>
#include <hip/hip_bf16.h>

#define H      768
#define H2     384
#define Bsz    32
#define NO     100
#define NT     50
#define LQ     20
#define EDGEF  5
#define LN_EPS 1e-5f

typedef __attribute__((ext_vector_type(16))) _Float16 v16h;
typedef __attribute__((ext_vector_type(8)))  _Float16 v8h;
typedef __attribute__((ext_vector_type(8)))  float    v8f;

// ---------------- WMMA fragment helpers (CDNA5 16x16x32 f16 layouts) -------

// A-matrix 16x32 (MxK), f16, from LDS row-major [.][stride].
// lane m = lane&15, laneHi = lane>>4. elem i<8: k=k0+8*hh+i ; i>=8: k=k0+16+8*hh+(i-8)
// -> two contiguous 8-half (16B) chunks at k0+8*hh and k0+16+8*hh.
__device__ inline v16h ldA(const _Float16* base, int stride, int m0, int k0) {
  const int lane = threadIdx.x & 31;
  const int hh   = lane >> 4;
  const _Float16* row = base + (size_t)(m0 + (lane & 15)) * stride + k0 + hh * 8;
  const v8h lo = *(const v8h*)(row);
  const v8h hi = *(const v8h*)(row + 16);
  v16h a;
#pragma unroll
  for (int i = 0; i < 8; ++i) { a[i] = lo[i]; a[8 + i] = hi[i]; }
  return a;
}

// B-matrix 32x16 (KxN) fragment from pre-tiled weights:
// Wt[((nt*KT + kt)*32 + lane)*16 + i]  == W[kt*32 + 8*(lane>>4) + (i&7) + 16*(i>>3)][nt*16 + (lane&15)]
// Each lane reads 32 contiguous bytes -> 2x global_load_b128, wave-coalesced.
__device__ inline v16h ldBt(const _Float16* Wt, int KT, int kt, int nt) {
  const int lane = threadIdx.x & 31;
  const _Float16* p = Wt + (((size_t)(nt * KT + kt) * 32) + lane) * 16;
  return *(const v16h*)p;
}

__device__ inline v8f wmma_f16(v16h a, v16h b, v8f c) {
  return __builtin_amdgcn_wmma_f32_16x16x32_f16(false, a, false, b, (short)0, c,
                                                false, false);
}

__device__ inline float wave_sum(float v) {
#pragma unroll
  for (int o = 16; o > 0; o >>= 1) v += __shfl_xor(v, o, 32);
  return v;
}
__device__ inline float wave_max(float v) {
#pragma unroll
  for (int o = 16; o > 0; o >>= 1) v = fmaxf(v, __shfl_xor(v, o, 32));
  return v;
}

// ---------------- prep: repack f32 [K][N] row-major -> WMMA-B tiled f16 ----

__global__ void k_repackB(const float* __restrict__ W, _Float16* __restrict__ Wt,
                          int K, int N) {
  const int KT = K / 32;
  const int total = K * N;
  for (int idx = blockIdx.x * 256 + threadIdx.x; idx < total; idx += gridDim.x * 256) {
    const int t    = idx >> 9;      // 512 elems per 32x16 tile
    const int r    = idx & 511;
    const int lane = r >> 4;
    const int i    = r & 15;
    const int nt   = t / KT, kt = t - nt * KT;
    const int k    = kt * 32 + ((lane >> 4) << 3) + (i & 7) + ((i >> 3) << 4);
    const int n    = nt * 16 + (lane & 15);
    Wt[idx] = (_Float16)W[(size_t)k * N + n];
  }
}

// ---------------- small kernels --------------------------------------------

// LayerNorm of q_emb -> q (first output region). One block per (b,l) row.
__global__ void k_qln(const float* __restrict__ x_in, const float* __restrict__ g,
                      const float* __restrict__ bb, float* __restrict__ y_out) {
  __shared__ float red[16];
  const int row = blockIdx.x;
  const float* x = x_in + (size_t)row * H;
  float* y = y_out + (size_t)row * H;
  const int t = threadIdx.x;
  float v0 = x[t], v1 = x[t + 256], v2 = x[t + 512];
  float s = v0 + v1 + v2, s2 = v0 * v0 + v1 * v1 + v2 * v2;
  s = wave_sum(s); s2 = wave_sum(s2);
  const int wave = t >> 5, lane = t & 31;
  if (lane == 0) { red[wave] = s; red[8 + wave] = s2; }
  __syncthreads();
  if (t == 0) {
    float ts = 0.f, ts2 = 0.f;
    for (int w = 0; w < 8; ++w) { ts += red[w]; ts2 += red[8 + w]; }
    red[0] = ts; red[8] = ts2;
  }
  __syncthreads();
  const float mean = red[0] * (1.f / H);
  const float var  = red[8] * (1.f / H) - mean * mean;
  const float inv  = rsqrtf(var + LN_EPS);
  y[t]       = (v0 - mean) * inv * g[t]       + bb[t];
  y[t + 256] = (v1 - mean) * inv * g[t + 256] + bb[t + 256];
  y[t + 512] = (v2 - mean) * inv * g[t + 512] + bb[t + 512];
}

// question pooled feature qf[b] = emb( sum_l attn_l * q[b,l,:] ), one block per (mod,b)
__global__ void k_qf(const float* __restrict__ q, const float* __restrict__ qmask,
                     const float* sw0, const float* sb0, const float* ew0, const float* eb0,
                     const float* sw1, const float* sb1, const float* ew1, const float* eb1,
                     float* qf0, float* qf1) {
  const int b   = blockIdx.x & (Bsz - 1);
  const int mod = blockIdx.x >> 5;
  const float* sw = mod ? sw1 : sw0;  const float* sb = mod ? sb1 : sb0;
  const float* ew = mod ? ew1 : ew0;  const float* eb = mod ? eb1 : eb0;
  float* qf = mod ? qf1 : qf0;
  __shared__ float attn[LQ];
  __shared__ float pooled[H];
  const float* qb = q + (size_t)b * LQ * H;
  const int wave = threadIdx.x >> 5, lane = threadIdx.x & 31;
  for (int l = wave; l < LQ; l += 8) {
    float acc = 0.f;
    for (int h = lane; h < H; h += 32) acc += qb[l * H + h] * sw[h];
    acc = wave_sum(acc);
    if (lane == 0) attn[l] = acc + sb[0];
  }
  __syncthreads();
  if (threadIdx.x == 0) {
    float mx = -1e30f;
    for (int l = 0; l < LQ; ++l) mx = fmaxf(mx, attn[l]);
    float z = 0.f;
    for (int l = 0; l < LQ; ++l) { float e = expf(attn[l] - mx) * qmask[b * LQ + l]; attn[l] = e; z += e; }
    const float iz = 1.f / z;
    for (int l = 0; l < LQ; ++l) attn[l] *= iz;
  }
  __syncthreads();
  for (int h = threadIdx.x; h < H; h += 256) {
    float acc = 0.f;
    for (int l = 0; l < LQ; ++l) acc += attn[l] * qb[l * H + h];
    pooled[h] = acc;
  }
  __syncthreads();
  for (int h = threadIdx.x; h < H; h += 256) {
    float acc = eb[h];
    for (int k = 0; k < H; ++k) acc += pooled[k] * ew[k * H + h];
    qf[b * H + h] = acc;
  }
}

// ---------------- fused edge-graph kernel (the hot one) --------------------
// one block per (b,o): edge MLP (layer1 VALU, layer2 WMMA into LDS as f16),
// tanh-score + masked softmax over s, A-weighted aggregation -> upd.
template <int O, int S>
__global__ void __launch_bounds__(256, 1) k_graph(
    const float* __restrict__ edge,                         // [B][O][S][5]
    const float* __restrict__ w1, const float* __restrict__ b1,   // [5][384],[384]
    const _Float16* __restrict__ w2t, const float* __restrict__ b2,// tiled [384x768] f16,[768]
    const float* __restrict__ qf,                           // [B][768]
    const float* __restrict__ eaw, const float* __restrict__ eab, // [768],[1]
    const float* __restrict__ omask,                        // [B][O]
    float* __restrict__ A_out,                              // [B][O][S]
    float* __restrict__ upd) {                              // [B][O][768]
  constexpr int Mpad   = (S + 15) & ~15;
  constexpr int MTILES = Mpad / 16;
  constexpr int h1st   = H2 + 8;
  constexpr int est    = H + 8;
  constexpr int KT     = H2 / 32;

  extern __shared__ unsigned char smem[];
  float* qf_s    = (float*)smem;            // 768
  float* w1_s    = qf_s + H;                // 1920
  float* b1_s    = w1_s + EDGEF * H2;       // 384
  float* b2_s    = b1_s + H2;               // 768
  float* eaw_s   = b2_s + H;                // 768
  float* edge_s  = eaw_s + H;               // 512
  float* score_s = edge_s + 512;            // 128
  float* a_s     = score_s + 128;           // 128
  _Float16* h1_s = (_Float16*)(a_s + 128);  // [Mpad][H2+8]
  _Float16* e_s  = h1_s + Mpad * h1st;      // [Mpad][H+8]

  const int b = blockIdx.x / O;
  const int o = blockIdx.x - b * O;
  const int tid = threadIdx.x;

  for (int i = tid; i < H; i += 256) { qf_s[i] = qf[b * H + i]; b2_s[i] = b2[i]; eaw_s[i] = eaw[i]; }
  for (int i = tid; i < EDGEF * H2; i += 256) w1_s[i] = w1[i];
  for (int i = tid; i < H2; i += 256) b1_s[i] = b1[i];
  const float* ep = edge + ((size_t)b * O + o) * S * EDGEF;
  for (int i = tid; i < S * EDGEF; i += 256) edge_s[i] = ep[i];
  __syncthreads();

  // layer1 (5->384) + ELU, zero-pad rows [S,Mpad)
  for (int i = tid; i < Mpad * H2; i += 256) {
    const int s = i / H2, k = i - s * H2;
    float v = 0.f;
    if (s < S) {
      v = b1_s[k];
      const float* er = edge_s + s * EDGEF;
#pragma unroll
      for (int j = 0; j < EDGEF; ++j) v += er[j] * w1_s[j * H2 + k];
      v = v > 0.f ? v : expm1f(v);
    }
    h1_s[s * h1st + k] = (_Float16)v;
  }
  __syncthreads();

  // layer2: e = h1 @ w2 + b2   (WMMA f16 -> f32 acc; acc fully in VGPRs)
  const int wave = tid >> 5, lane = tid & 31, hh = lane >> 4;
  for (int nt = wave; nt < H / 16; nt += 8) {
    const int n0 = nt * 16;
    v8f acc[MTILES];
#pragma unroll
    for (int mt = 0; mt < MTILES; ++mt)
#pragma unroll
      for (int r = 0; r < 8; ++r) acc[mt][r] = 0.f;
    for (int kt = 0; kt < KT; ++kt) {
      const v16h bf = ldBt(w2t, KT, kt, nt);
#pragma unroll
      for (int mt = 0; mt < MTILES; ++mt) {
        const v16h af = ldA(h1_s, h1st, mt * 16, kt * 32);
        acc[mt] = wmma_f16(af, bf, acc[mt]);
      }
    }
    const int n = n0 + (lane & 15);
    const float bias = b2_s[n];
#pragma unroll
    for (int mt = 0; mt < MTILES; ++mt)
#pragma unroll
      for (int r = 0; r < 8; ++r) {
        const int m = mt * 16 + r + hh * 8;
        e_s[m * est + n] = (_Float16)(acc[mt][r] + bias);
      }
  }
  __syncthreads();

  // score[s] = ea_b + sum_h ea_w[h] * tanh(qf[h] + e[s,h])
  for (int s = wave; s < S; s += 8) {
    float acc = 0.f;
    const _Float16* er = e_s + s * est;
    for (int h = lane; h < H; h += 32) acc += eaw_s[h] * tanhf(qf_s[h] + (float)er[h]);
    acc = wave_sum(acc);
    if (lane == 0) score_s[s] = acc + eab[0];
  }
  __syncthreads();

  // masked softmax over s;  A = softmax * m / (m + 1e-12)
  if (wave == 0) {
    float mx = -1e30f;
    for (int s = lane; s < S; s += 32) mx = fmaxf(mx, score_s[s]);
    mx = wave_max(mx);
    float z = 0.f;
    for (int s = lane; s < S; s += 32) { const float e = expf(score_s[s] - mx); a_s[s] = e; z += e; }
    z = wave_sum(z);
    const float m = omask[b * O + o];
    const float scale = (m / (m + 1e-12f)) / z;
    float* Ar = A_out + ((size_t)b * O + o) * S;
    for (int s = lane; s < S; s += 32) { const float a = a_s[s] * scale; a_s[s] = a; Ar[s] = a; }
  }
  __syncthreads();

  // upd[h] = sum_s a_s[s] * e[s,h]
  float* ur = upd + ((size_t)b * O + o) * H;
  for (int h = tid; h < H; h += 256) {
    float acc = 0.f;
#pragma unroll 5
    for (int s = 0; s < S; ++s) acc += a_s[s] * (float)e_s[s * est + h];
    ur[h] = acc;
  }
}

// out[b,m,:] = maskM[b,m] * sum_k A[b,k,m]*maskK[b,k]*feat[b,k,:]
__global__ void k_attnapply(const float* __restrict__ A, const float* __restrict__ feat,
                            const float* __restrict__ maskK, const float* __restrict__ maskM,
                            float* __restrict__ out, int K, int M) {
  __shared__ float w_s[128];
  const int b = blockIdx.x / M;
  const int m = blockIdx.x - b * M;
  for (int k = threadIdx.x; k < K; k += 256)
    w_s[k] = A[((size_t)b * K + k) * M + m] * maskK[b * K + k];
  __syncthreads();
  const float mm = maskM[b * M + m];
  const float* fb = feat + (size_t)b * K * H;
  float* ob = out + ((size_t)b * M + m) * H;
  for (int h = threadIdx.x; h < H; h += 256) {
    float acc = 0.f;
    for (int k = 0; k < K; ++k) acc += w_s[k] * fb[(size_t)k * H + h];
    ob[h] = acc * mm;
  }
}

// out = LN(x0@W0+b0) + LN(x1@W1+b1) + LN(x2@W2+b2); 16 rows per block, WMMA GEMMs
__global__ void __launch_bounds__(256, 1) k_featsum(
    const float* x0, const float* x1, const float* x2,
    const _Float16* w0, const _Float16* w1, const _Float16* w2,   // tiled
    const float* lb0, const float* lb1, const float* lb2,
    const float* g0, const float* g1, const float* g2,
    const float* be0, const float* be1, const float* be2,
    float* __restrict__ out) {
  constexpr int xst = H + 8;
  constexpr int rst = H + 4;
  constexpr int KT  = H / 32;
  extern __shared__ unsigned char smem[];
  _Float16* X = (_Float16*)smem;             // [16][H+8]
  float* res   = (float*)(X + 16 * xst);     // [16][H+4]
  float* accum = res + 16 * rst;             // [16][H]

  const float*    xs[3]  = {x0, x1, x2};
  const _Float16* wsp[3] = {w0, w1, w2};
  const float*    lbs[3] = {lb0, lb1, lb2};
  const float*    gs[3]  = {g0, g1, g2};
  const float*    bes[3] = {be0, be1, be2};

  const int row0 = blockIdx.x * 16;
  const int tid = threadIdx.x, wave = tid >> 5, lane = tid & 31, hh = lane >> 4;

  for (int j = 0; j < 3; ++j) {
    for (int i = tid; i < 16 * H; i += 256) {
      const int r = i / H, k = i - r * H;
      X[r * xst + k] = (_Float16)xs[j][(size_t)(row0 + r) * H + k];
    }
    __syncthreads();
    for (int nt = wave; nt < H / 16; nt += 8) {
      const int n0 = nt * 16;
      v8f acc;
#pragma unroll
      for (int r = 0; r < 8; ++r) acc[r] = 0.f;
      for (int kt = 0; kt < KT; ++kt) {
        const v16h bf = ldBt(wsp[j], KT, kt, nt);
        const v16h af = ldA(X, xst, 0, kt * 32);
        acc = wmma_f16(af, bf, acc);
      }
      const int n = n0 + (lane & 15);
      const float bias = lbs[j][n];
#pragma unroll
      for (int r = 0; r < 8; ++r) res[(r + hh * 8) * rst + n] = acc[r] + bias;
    }
    __syncthreads();
    for (int rr = 0; rr < 2; ++rr) {
      const int r = wave * 2 + rr;
      float s = 0.f, s2 = 0.f;
      for (int h = lane; h < H; h += 32) { const float v = res[r * rst + h]; s += v; s2 += v * v; }
      s = wave_sum(s); s2 = wave_sum(s2);
      const float mean = s * (1.f / H);
      const float var  = s2 * (1.f / H) - mean * mean;
      const float inv  = rsqrtf(var + LN_EPS);
      for (int h = lane; h < H; h += 32) {
        const float v = (res[r * rst + h] - mean) * inv * gs[j][h] + bes[j][h];
        if (j == 0) accum[r * H + h] = v; else accum[r * H + h] += v;
      }
    }
    __syncthreads();
  }
  float* ob = out + (size_t)row0 * H;
  for (int i = tid; i < 16 * H; i += 256) ob[i] = accum[i];
}

// ---------------- host launcher --------------------------------------------

extern "C" void kernel_launch(void* const* d_in, const int* in_sizes, int n_in,
                              void* d_out, int out_size, void* d_ws, size_t ws_size,
                              hipStream_t stream) {
  (void)in_sizes; (void)n_in; (void)out_size; (void)ws_size;
  const float* v_feat   = (const float*)d_in[0];
  const float* t_feat   = (const float*)d_in[1];
  const float* v2t_edge = (const float*)d_in[2];
  const float* t2v_edge = (const float*)d_in[3];
  const float* q_emb    = (const float*)d_in[4];
  const float* v_mask   = (const float*)d_in[5];
  const float* t_mask   = (const float*)d_in[6];
  const float* q_mask   = (const float*)d_in[7];
  const float* qn_b     = (const float*)d_in[8];   // pytree: dict keys sorted
  const float* qn_g     = (const float*)d_in[9];

  // module leaves in alphabetical (jax pytree) order
  enum { L_ea_b = 0, L_ea_w, L_emb_b, L_emb_w, L_f1_b, L_f1_be, L_f1_g, L_f1_w,
         L_f2_b, L_f2_be, L_f2_g, L_f2_w, L_f3_b, L_f3_be, L_f3_g, L_f3_w,
         L_self_b, L_self_w, L_te_b1, L_te_b2, L_te_w1, L_te_w2 };
  const int QT = 10, QV = 32;  // 'qt' < 'qv'
  auto P = [&](int base, int leaf) { return (const float*)d_in[base + leaf]; };

  // workspace carve-up
  unsigned char* wsb = (unsigned char*)d_ws;
  size_t off = 0;
  auto take = [&](size_t bytes) -> void* {
    void* p = wsb + off; off = (off + bytes + 255) & ~(size_t)255; return p;
  };
  _Float16* w2qv = (_Float16*)take((size_t)H2 * H * 2);
  _Float16* w2qt = (_Float16*)take((size_t)H2 * H * 2);
  _Float16* fwqv[3]; _Float16* fwqt[3];
  for (int j = 0; j < 3; ++j) fwqv[j] = (_Float16*)take((size_t)H * H * 2);
  for (int j = 0; j < 3; ++j) fwqt[j] = (_Float16*)take((size_t)H * H * 2);
  float* qf_qv = (float*)take((size_t)Bsz * H * 4);
  float* qf_qt = (float*)take((size_t)Bsz * H * 4);
  float* A_v2t = (float*)take((size_t)Bsz * NO * NT * 4);
  float* A_t2v = (float*)take((size_t)Bsz * NT * NO * 4);
  float* upd_v = (float*)take((size_t)Bsz * NO * H * 4);
  float* upd_t = (float*)take((size_t)Bsz * NT * H * 4);
  float* new_v = (float*)take((size_t)Bsz * NO * H * 4);
  float* new_t = (float*)take((size_t)Bsz * NT * H * 4);

  float* q_out = (float*)d_out;
  float* v_out = q_out + (size_t)Bsz * LQ * H;
  float* t_out = v_out + (size_t)Bsz * NO * H;

  // weight f32 -> f16 repack into WMMA-B tile order (L2-resident afterwards)
  k_repackB<<<288, 256, 0, stream>>>(P(QV, L_te_w2), w2qv, H2, H);
  k_repackB<<<288, 256, 0, stream>>>(P(QT, L_te_w2), w2qt, H2, H);
  const int fwleaf[3] = {L_f1_w, L_f2_w, L_f3_w};
  for (int j = 0; j < 3; ++j) {
    k_repackB<<<576, 256, 0, stream>>>(P(QV, fwleaf[j]), fwqv[j], H, H);
    k_repackB<<<576, 256, 0, stream>>>(P(QT, fwleaf[j]), fwqt[j], H, H);
  }

  // q = LN(q_emb)
  k_qln<<<Bsz * LQ, 256, 0, stream>>>(q_emb, qn_g, qn_b, q_out);

  // pooled question features for both modules
  k_qf<<<2 * Bsz, 256, 0, stream>>>(q_out, q_mask,
      P(QV, L_self_w), P(QV, L_self_b), P(QV, L_emb_w), P(QV, L_emb_b),
      P(QT, L_self_w), P(QT, L_self_b), P(QT, L_emb_w), P(QT, L_emb_b),
      qf_qv, qf_qt);

  auto graph_lds = [](int S) -> size_t {
    const int Mpad = (S + 15) & ~15;
    return (size_t)(H + EDGEF * H2 + H2 + H + H + 512 + 128 + 128) * 4
         + (size_t)Mpad * (H2 + 8) * 2 + (size_t)Mpad * (H + 8) * 2;
  };
  // qv: edges [B,NO,NT,5], O=NO, S=NT  (LDS ~171 KB)
  k_graph<NO, NT><<<Bsz * NO, 256, graph_lds(NT), stream>>>(v2t_edge,
      P(QV, L_te_w1), P(QV, L_te_b1), w2qv, P(QV, L_te_b2),
      qf_qv, P(QV, L_ea_w), P(QV, L_ea_b), v_mask, A_v2t, upd_v);
  // qt: edges [B,NT,NO,5], O=NT, S=NO  (LDS ~283 KB, fits 320 KB WGP)
  k_graph<NT, NO><<<Bsz * NT, 256, graph_lds(NO), stream>>>(t2v_edge,
      P(QT, L_te_w1), P(QT, L_te_b1), w2qt, P(QT, L_te_b2),
      qf_qt, P(QT, L_ea_w), P(QT, L_ea_b), t_mask, A_t2v, upd_t);

  // new_t[b,t,:] = sum_o A_v2t[b,o,t]*vm[b,o]*tm[b,t]*v_feat[b,o,:]
  k_attnapply<<<Bsz * NT, 256, 0, stream>>>(A_v2t, v_feat, v_mask, t_mask, new_t, NO, NT);
  // new_v[b,o,:] = sum_t A_t2v[b,t,o]*tm[b,t]*vm[b,o]*t_feat[b,t,:]
  k_attnapply<<<Bsz * NO, 256, 0, stream>>>(A_t2v, t_feat, t_mask, v_mask, new_v, NT, NO);

  const size_t fsLds = (size_t)16 * (H + 8) * 2 + (size_t)16 * (H + 4) * 4 + (size_t)16 * H * 4;
  k_featsum<<<(Bsz * NO) / 16, 256, fsLds, stream>>>(v_feat, new_v, upd_v,
      fwqv[0], fwqv[1], fwqv[2],
      P(QV, L_f1_b), P(QV, L_f2_b), P(QV, L_f3_b),
      P(QV, L_f1_g), P(QV, L_f2_g), P(QV, L_f3_g),
      P(QV, L_f1_be), P(QV, L_f2_be), P(QV, L_f3_be),
      v_out);
  k_featsum<<<(Bsz * NT) / 16, 256, fsLds, stream>>>(t_feat, new_t, upd_t,
      fwqt[0], fwqt[1], fwqt[2],
      P(QT, L_f1_b), P(QT, L_f2_b), P(QT, L_f3_b),
      P(QT, L_f1_g), P(QT, L_f2_g), P(QT, L_f3_g),
      P(QT, L_f1_be), P(QT, L_f2_be), P(QT, L_f3_be),
      t_out);
}
